// W8A8LinearDynamic_61409442398840
// MI455X (gfx1250) — 8-bit matrix operand order repaired and checked by reading, compiled, not run on hardware
//
#include <hip/hip_runtime.h>

typedef __attribute__((ext_vector_type(8))) int v8i;

#define TOK 4096
#define IN  4096
#define OUT 12288

#define BM 128
#define BN 128
#define BK 128
#define LDS_STRIDE 144

#if defined(__has_builtin)
#if __has_builtin(__builtin_amdgcn_global_load_async_to_lds_b128)
#define HAVE_ASYNC_LDS 1
#endif
#endif
#ifndef HAVE_ASYNC_LDS
#define HAVE_ASYNC_LDS 0
#endif

typedef int v4i_b128 __attribute__((vector_size(16)));
typedef __attribute__((address_space(1))) v4i_b128* gptr_b128;
typedef __attribute__((address_space(3))) v4i_b128* lptr_b128;

__device__ __forceinline__ void copy16_to_lds(const signed char* g, unsigned char* l) {
#if HAVE_ASYNC_LDS
  __builtin_amdgcn_global_load_async_to_lds_b128(
      (gptr_b128)(void*)g, (lptr_b128)(void*)l,  0,  0);
#else
  *(int4*)l = *(const int4*)g;
#endif
}

__global__ void w8a8_quant_kernel(const float* __restrict__ x,
                                  signed char* __restrict__ xq,
                                  float* __restrict__ act_scale) {
  __shared__ float red[8];
  __shared__ float sbc;
  const int t   = blockIdx.x;
  const int tid = threadIdx.x;
  const int wave = tid >> 5;
  const int lane = tid & 31;

  const float* row = x + (size_t)t * IN;
  const float4* row4 = (const float4*)row;

  float4 v[4];
  float m = 0.0f;
#pragma unroll
  for (int i = 0; i < 4; ++i) {
    v[i] = row4[tid * 4 + i];
    m = fmaxf(m, fabsf(v[i].x));
    m = fmaxf(m, fabsf(v[i].y));
    m = fmaxf(m, fabsf(v[i].z));
    m = fmaxf(m, fabsf(v[i].w));
  }
#pragma unroll
  for (int off = 16; off > 0; off >>= 1)
    m = fmaxf(m, __shfl_xor(m, off, 32));
  if (lane == 0) red[wave] = m;
  __syncthreads();
  if (tid == 0) {
    float mm = red[0];
#pragma unroll
    for (int i = 1; i < 8; ++i) mm = fmaxf(mm, red[i]);
    float s = fmaxf(mm * (1.0f / 127.0f), 1e-8f);
    act_scale[t] = s;
    sbc = s;
  }
  __syncthreads();
  const float s = sbc;

  int p[4];
#pragma unroll
  for (int i = 0; i < 4; ++i) {
    int a = (int)fminf(fmaxf(__builtin_rintf(v[i].x / s), -128.0f), 127.0f);
    int b = (int)fminf(fmaxf(__builtin_rintf(v[i].y / s), -128.0f), 127.0f);
    int c = (int)fminf(fmaxf(__builtin_rintf(v[i].z / s), -128.0f), 127.0f);
    int d = (int)fminf(fmaxf(__builtin_rintf(v[i].w / s), -128.0f), 127.0f);
    p[i] = (a & 255) | ((b & 255) << 8) | ((c & 255) << 16) | ((d & 255) << 24);
  }
  int4* dst = (int4*)(xq + (size_t)t * IN);
  dst[tid] = make_int4(p[0], p[1], p[2], p[3]);
}

__global__ void __launch_bounds__(256)
w8a8_wmma_gemm_kernel(const signed char* __restrict__ xq,
                      const signed char* __restrict__ weight,
                      const float* __restrict__ act_scale,
                      const float* __restrict__ weight_scale,
                      const float* __restrict__ bias,
                      float* __restrict__ out) {
  __shared__ __attribute__((aligned(16))) unsigned char smemA[2][BM * LDS_STRIDE];
  __shared__ __attribute__((aligned(16))) unsigned char smemB[2][BN * LDS_STRIDE];

  const int tid  = threadIdx.x;
  const int lane = tid & 31;
  const int half = lane >> 4;
  const int l16  = lane & 15;
  const int wave = tid >> 5;
  const int wm   = wave & 1;
  const int wn   = wave >> 1;

  const int n0 = blockIdx.x * BN;
  const int m0 = blockIdx.y * BM;

  v8i acc[4][2];
#pragma unroll
  for (int mi = 0; mi < 4; ++mi)
#pragma unroll
    for (int ni = 0; ni < 2; ++ni)
      acc[mi][ni] = (v8i)(0);

  const signed char* aBase = xq + (size_t)m0 * IN;
  const signed char* bBase = weight + (size_t)n0 * IN;

  const int crow[4] = { (tid + 0)   >> 3, (tid + 256) >> 3,
                        (tid + 512) >> 3, (tid + 768) >> 3 };
  const int ccol = (tid & 7) << 4;

#pragma unroll
  for (int i = 0; i < 4; ++i) {
    copy16_to_lds(aBase + (size_t)crow[i] * IN + ccol,
                  &smemA[0][crow[i] * LDS_STRIDE + ccol]);
    copy16_to_lds(bBase + (size_t)crow[i] * IN + ccol,
                  &smemB[0][crow[i] * LDS_STRIDE + ccol]);
  }

  int cur = 0;
  for (int k0 = 0; k0 < IN; k0 += BK) {
    const bool hasNext = (k0 + BK) < IN;

    if (hasNext) {
      const signed char* aSrc = aBase + k0 + BK;
      const signed char* bSrc = bBase + k0 + BK;
#pragma unroll
      for (int i = 0; i < 4; ++i) {
        copy16_to_lds(aSrc + (size_t)crow[i] * IN + ccol,
                      &smemA[cur ^ 1][crow[i] * LDS_STRIDE + ccol]);
        copy16_to_lds(bSrc + (size_t)crow[i] * IN + ccol,
                      &smemB[cur ^ 1][crow[i] * LDS_STRIDE + ccol]);
      }
      if (k0 + 2 * BK < IN) {
        __builtin_prefetch(aBase + (size_t)(tid >> 1) * IN + k0 + 2 * BK + (tid & 1) * 64, 0, 1);
        __builtin_prefetch(bBase + (size_t)(tid >> 1) * IN + k0 + 2 * BK + (tid & 1) * 64, 0, 1);
      }
    }

#if HAVE_ASYNC_LDS
    if (hasNext) asm volatile("s_wait_asynccnt 8" ::: "memory");
    else         asm volatile("s_wait_asynccnt 0" ::: "memory");
#endif
    __syncthreads();

    const unsigned char* sA = smemA[cur];
    const unsigned char* sB = smemB[cur];

#pragma unroll
    for (int kk = 0; kk < BK; kk += 64) {
      v8i afrag[4];
#pragma unroll
      for (int mi = 0; mi < 4; ++mi) {
        const unsigned char* ap =
            sA + (wm * 64 + mi * 16 + l16) * LDS_STRIDE + kk + 8 * half;
        int2 p0 = *(const int2*)(ap);
        int2 p1 = *(const int2*)(ap + 16);
        int2 p2 = *(const int2*)(ap + 32);
        int2 p3 = *(const int2*)(ap + 48);
        afrag[mi] = (v8i){p0.x, p0.y, p1.x, p1.y, p2.x, p2.y, p3.x, p3.y};
      }
#pragma unroll
      for (int ni = 0; ni < 2; ++ni) {
        const unsigned char* bp =
            sB + (wn * 32 + ni * 16 + l16) * LDS_STRIDE + kk + 8 * half;
        int2 q0 = *(const int2*)(bp);
        int2 q1 = *(const int2*)(bp + 16);
        int2 q2 = *(const int2*)(bp + 32);
        int2 q3 = *(const int2*)(bp + 48);
        v8i bfrag = (v8i){q0.x, q0.y, q1.x, q1.y, q2.x, q2.y, q3.x, q3.y};
#pragma unroll
        for (int mi = 0; mi < 4; ++mi) {
          acc[mi][ni] = __builtin_amdgcn_wmma_i32_16x16x64_iu8(
              true, afrag[mi], true, bfrag, acc[mi][ni],
               false,  false);
        }
      }
    }

    __syncthreads();
    cur ^= 1;
  }

#pragma unroll
  for (int ni = 0; ni < 2; ++ni) {
    const int n = n0 + wn * 32 + ni * 16 + l16;
    const float ws = weight_scale[n];
    const float bs = bias[n];
#pragma unroll
    for (int mi = 0; mi < 4; ++mi) {
#pragma unroll
      for (int r = 0; r < 8; ++r) {
        const int m = m0 + wm * 64 + mi * 16 + r + 8 * half;
        out[(size_t)m * OUT + n] =
            (float)acc[mi][ni][r] * act_scale[m] * ws + bs;
      }
    }
  }
}

extern "C" void kernel_launch(void* const* d_in, const int* in_sizes, int n_in,
                              void* d_out, int out_size, void* d_ws, size_t ws_size,
                              hipStream_t stream) {
  const float*       x            = (const float*)d_in[0];
  const signed char* weight       = (const signed char*)d_in[1];
  const float*       weight_scale = (const float*)d_in[2];
  const float*       bias         = (const float*)d_in[3];
  float*             out          = (float*)d_out;

  signed char* xq     = (signed char*)d_ws;
  float*       ascale = (float*)((char*)d_ws + (size_t)TOK * IN);

  w8a8_quant_kernel<<<TOK, 256, 0, stream>>>(x, xq, ascale);

  dim3 grid(OUT / BN, TOK / BM);
  w8a8_wmma_gemm_kernel<<<grid, 256, 0, stream>>>(xq, weight, ascale,
                                                  weight_scale, bias, out);
}
